// _MatrixDecomposition1DBase_19043884991170
// MI455X (gfx1250) — matrix operand order repaired and checked by reading, not compiled, not run on hardware
//
#include <hip/hip_runtime.h>

// Problem constants (from reference): B=32, D=512, N=4096, R=64, 7 steps.
#define BB 32
#define DD 512
#define NN 4096
#define RR 64
#define EPSV 1e-6f

typedef __attribute__((ext_vector_type(16))) __bf16 v16bf;
typedef __attribute__((ext_vector_type(8)))  __bf16 v8bf;
typedef __attribute__((ext_vector_type(8)))  float  v8f;

__device__ __forceinline__ v8f wmma_bf16(v16bf a, v16bf b, v8f c) {
  // D(f32 16x16) = A(bf16 16x32) * B(bf16 32x16) + C
  return __builtin_amdgcn_wmma_f32_16x16x32_bf16(false, a, false, b, (short)0, c,
                                                 false, false);
}

// ---- Fragment loaders -------------------------------------------------------
// A fragment (16x32, 16-bit): lane holds row `row`; element e maps to
// k = (e&7) + (e>=8?16:0) + (lane>=16?8:0)  -> two contiguous 8-elem runs.
__device__ __forceinline__ v16bf load_a_bf16(const __bf16* p, int row, int ld,
                                             int k0, int lane) {
  const int h = (lane >> 4) & 1;
  const __bf16* base = p + (size_t)row * ld + k0 + h * 8;
  v8bf lo = *(const v8bf*)base;
  v8bf hi = *(const v8bf*)(base + 16);
  v16bf a;
#pragma unroll
  for (int i = 0; i < 8; ++i) { a[i] = lo[i]; a[i + 8] = hi[i]; }
  return a;
}

__device__ __forceinline__ v16bf load_a_f32(const float* p, int row, int ld,
                                            int k0, int lane) {
  const int h = (lane >> 4) & 1;
  const float* base = p + (size_t)row * ld + k0 + h * 8;
  v16bf a;
#pragma unroll
  for (int i = 0; i < 8; ++i) {
    a[i]     = (__bf16)base[i];
    a[i + 8] = (__bf16)base[i + 16];
  }
  return a;
}

// B fragment (32x16, 16-bit) from TRANSPOSED storage srcT[col][k] (stride ld):
// lane holds column col0+(lane&15); k = (e&7) + (e>=8?16:0) + (lane>=16?8:0).
__device__ __forceinline__ v16bf load_bT_bf16(const __bf16* p, int col0, int ld,
                                              int k0, int lane) {
  const int h = (lane >> 4) & 1;
  const __bf16* base = p + (size_t)(col0 + (lane & 15)) * ld + k0 + h * 8;
  v8bf lo = *(const v8bf*)base;
  v8bf hi = *(const v8bf*)(base + 16);
  v16bf b;
#pragma unroll
  for (int i = 0; i < 8; ++i) { b[i] = lo[i]; b[i + 8] = hi[i]; }
  return b;
}

__device__ __forceinline__ v16bf load_bT_f32(const float* p, int col0, int ld,
                                             int k0, int lane) {
  const int h = (lane >> 4) & 1;
  const float* base = p + (size_t)(col0 + (lane & 15)) * ld + k0 + h * 8;
  v16bf b;
#pragma unroll
  for (int i = 0; i < 16; ++i) b[i] = (__bf16)base[i < 8 ? i : i + 8];
  return b;
}

// ---- Prep: XT = bf16(x^T) [B,N,D] via LDS-tiled transpose -------------------
__global__ void k_prep_x(const float* __restrict__ x, __bf16* __restrict__ XT) {
  __shared__ __bf16 t[64][65];
  const int b = blockIdx.z;
  const int n0 = blockIdx.x * 64, d0 = blockIdx.y * 64;
  const float* xb = x + (size_t)b * DD * NN;
  __bf16* XTb = XT + (size_t)b * NN * DD;
  const int tn = threadIdx.x & 63;   // fast index: n on load, d on store
  const int td = threadIdx.x >> 6;   // 0..3
#pragma unroll
  for (int i = 0; i < 64; i += 4) {
    float v = xb[(size_t)(d0 + td + i) * NN + n0 + tn];
    t[tn][td + i] = (__bf16)v;       // t[n][d]
  }
  __syncthreads();
#pragma unroll
  for (int i = 0; i < 64; i += 4)
    XTb[(size_t)(n0 + td + i) * DD + d0 + tn] = t[td + i][tn];
}

// ---- Prep: basesW (f32 working copy, inputs must not be mutated) + basesT bf16
__global__ void k_prep_bases(const float* __restrict__ bases,
                             float* __restrict__ basesW,
                             __bf16* __restrict__ basesT) {
  const int b = blockIdx.x;
  const float* src = bases + (size_t)b * DD * RR;
  float* dst = basesW + (size_t)b * DD * RR;
  __bf16* dt = basesT + (size_t)b * RR * DD;
  for (int i = threadIdx.x; i < DD * RR; i += blockDim.x) {
    float v = src[i];
    dst[i] = v;
    dt[(size_t)(i % RR) * DD + (i / RR)] = (__bf16)v;
  }
}

// ---- Gram: G = M^T M (64x64, symmetric) from MT [64,K] bf16; bf16 out -------
__global__ void k_gram(const __bf16* __restrict__ MT, int K,
                       __bf16* __restrict__ gram) {
  const int b = blockIdx.x;
  const int lane = threadIdx.x & 31, wave = threadIdx.x >> 5;
  const __bf16* Mb = MT + (size_t)b * RR * K;
  const int t0 = wave * 2;                        // 8 waves x 2 tiles = 16 tiles
  v8f acc[2] = {};
  for (int k0 = 0; k0 < K; k0 += 32) {
#pragma unroll
    for (int u = 0; u < 2; ++u) {
      const int t = t0 + u, ri = (t >> 2) * 16, rj = (t & 3) * 16;
      v16bf a = load_a_bf16(Mb, ri + (lane & 15), K, k0, lane);
      v16bf bb = load_bT_bf16(Mb, rj, K, k0, lane);
      acc[u] = wmma_bf16(a, bb, acc[u]);
    }
  }
  __bf16* g = gram + (size_t)b * RR * RR;
  const int h = (lane >> 4) & 1;
#pragma unroll
  for (int u = 0; u < 2; ++u) {
    const int t = t0 + u, ri = (t >> 2) * 16, rj = (t & 3) * 16;
#pragma unroll
    for (int i = 0; i < 8; ++i)
      g[(size_t)(ri + i + h * 8) * RR + rj + (lane & 15)] = (__bf16)acc[u][i];
  }
}

// ---- coef update: numer = XT*bases (WMMA K=512); mode0: softmax(numer);
//      mode1: coef *= numer / (coef@gramB + eps). Writes coef f32 + coefT bf16.
__global__ void k_coef_update(const __bf16* __restrict__ XT,
                              const __bf16* __restrict__ basesT,
                              const __bf16* __restrict__ gramB,
                              float* __restrict__ coef,
                              __bf16* __restrict__ coefT, int mode) {
  const int lane = threadIdx.x & 31, wave = threadIdx.x >> 5;
  const int b = blockIdx.y;
  const int n0 = blockIdx.x * 128 + wave * 16;    // 16 rows per wave
  const __bf16* XTb = XT + (size_t)b * NN * DD;
  const __bf16* bTb = basesT + (size_t)b * RR * DD;
  const int arow = n0 + (lane & 15);
  const int h = (lane >> 4) & 1;

  v8f acc[4] = {};
  for (int k0 = 0; k0 < DD; k0 += 32) {
    v16bf a = load_a_bf16(XTb, arow, DD, k0, lane);
    if (k0 + 64 < DD)
      __builtin_prefetch(XTb + (size_t)arow * DD + k0 + 64, 0, 0);
#pragma unroll
    for (int j = 0; j < 4; ++j)
      acc[j] = wmma_bf16(a, load_bT_bf16(bTb, j * 16, DD, k0, lane), acc[j]);
  }

  float* cb = coef + (size_t)b * NN * RR;
  if (mode == 0) {
    // softmax over r (64 vals/row): 4 in-lane + 16-lane half reduction
#pragma unroll
    for (int i = 0; i < 8; ++i) {
      float mx = acc[0][i];
#pragma unroll
      for (int j = 1; j < 4; ++j) mx = fmaxf(mx, acc[j][i]);
#pragma unroll
      for (int s = 1; s < 16; s <<= 1) mx = fmaxf(mx, __shfl_xor(mx, s, 32));
      float e[4], sum = 0.f;
#pragma unroll
      for (int j = 0; j < 4; ++j) { e[j] = __expf(acc[j][i] - mx); sum += e[j]; }
#pragma unroll
      for (int s = 1; s < 16; s <<= 1) sum += __shfl_xor(sum, s, 32);
      const float inv = 1.f / sum;
#pragma unroll
      for (int j = 0; j < 4; ++j) acc[j][i] = e[j] * inv;
    }
  } else {
    // denom = coef_old @ gramB (symmetric -> B frags from row-major gram)
    const __bf16* g = gramB + (size_t)b * RR * RR;
    v8f den[4] = {};
    for (int k0 = 0; k0 < RR; k0 += 32) {
      v16bf a = load_a_f32(cb, arow, RR, k0, lane);
#pragma unroll
      for (int j = 0; j < 4; ++j)
        den[j] = wmma_bf16(a, load_bT_bf16(g, j * 16, RR, k0, lane), den[j]);
    }
#pragma unroll
    for (int j = 0; j < 4; ++j)
#pragma unroll
      for (int i = 0; i < 8; ++i) {
        const int row = n0 + i + h * 8, col = j * 16 + (lane & 15);
        const float cold = cb[(size_t)row * RR + col];
        acc[j][i] = cold * acc[j][i] / (den[j][i] + EPSV);
      }
  }

  __bf16* ct = coefT + (size_t)b * RR * NN;
#pragma unroll
  for (int j = 0; j < 4; ++j)
#pragma unroll
    for (int i = 0; i < 8; ++i) {
      const int row = n0 + i + h * 8, col = j * 16 + (lane & 15);
      const float v = acc[j][i];
      cb[(size_t)row * RR + col] = v;
      ct[(size_t)col * NN + row] = (__bf16)v;
    }
}

// ---- bases update: numer = x*coef (WMMA K=4096, f32 x converted on load);
//      bases *= numer / (bases@gramC + eps). Writes basesW f32 + basesT bf16.
__global__ void k_bases_update(const float* __restrict__ x,
                               const __bf16* __restrict__ coefT,
                               const __bf16* __restrict__ gramC,
                               float* __restrict__ basesW,
                               __bf16* __restrict__ basesT) {
  const int lane = threadIdx.x & 31, wave = threadIdx.x >> 5;
  const int b = blockIdx.y;
  const int d0 = blockIdx.x * 128 + wave * 16;
  const float* xb = x + (size_t)b * DD * NN;
  const __bf16* ctb = coefT + (size_t)b * RR * NN;
  const int arow = d0 + (lane & 15);
  const int h = (lane >> 4) & 1;

  v8f acc[4] = {};
  for (int k0 = 0; k0 < NN; k0 += 32) {
    v16bf a = load_a_f32(xb, arow, NN, k0, lane);
    if (k0 + 64 < NN)
      __builtin_prefetch(xb + (size_t)arow * NN + k0 + 64, 0, 0);
#pragma unroll
    for (int j = 0; j < 4; ++j)
      acc[j] = wmma_bf16(a, load_bT_bf16(ctb, j * 16, NN, k0, lane), acc[j]);
  }

  float* bw = basesW + (size_t)b * DD * RR;
  const __bf16* g = gramC + (size_t)b * RR * RR;
  v8f den[4] = {};
  for (int k0 = 0; k0 < RR; k0 += 32) {
    v16bf a = load_a_f32(bw, arow, RR, k0, lane);
#pragma unroll
    for (int j = 0; j < 4; ++j)
      den[j] = wmma_bf16(a, load_bT_bf16(g, j * 16, RR, k0, lane), den[j]);
  }

  __bf16* bt = basesT + (size_t)b * RR * DD;
#pragma unroll
  for (int j = 0; j < 4; ++j)
#pragma unroll
    for (int i = 0; i < 8; ++i) {
      const int row = d0 + i + h * 8, col = j * 16 + (lane & 15);
      const float old = bw[(size_t)row * RR + col];
      const float v = old * acc[j][i] / (den[j][i] + EPSV);
      bw[(size_t)row * RR + col] = v;
      bt[(size_t)col * DD + row] = (__bf16)v;
    }
}

// ---- Reconstruction: out = bases @ coef^T  (D x N per batch, K = R = 64) ----
__global__ void k_recon(const float* __restrict__ basesW,
                        const float* __restrict__ coef,
                        float* __restrict__ out) {
  const int lane = threadIdx.x & 31, wave = threadIdx.x >> 5;
  const int b = blockIdx.z;
  const int d0 = blockIdx.y * 128 + wave * 16;
  const int n0 = blockIdx.x * 64;
  const float* bw = basesW + (size_t)b * DD * RR;
  const float* cf = coef + (size_t)b * NN * RR;
  float* ob = out + (size_t)b * DD * NN;
  const int h = (lane >> 4) & 1;

  v8f acc[4] = {};
  for (int k0 = 0; k0 < RR; k0 += 32) {
    v16bf a = load_a_f32(bw, d0 + (lane & 15), RR, k0, lane);
#pragma unroll
    for (int j = 0; j < 4; ++j)
      acc[j] = wmma_bf16(a, load_bT_f32(cf, n0 + j * 16, RR, k0, lane), acc[j]);
  }
#pragma unroll
  for (int j = 0; j < 4; ++j)
#pragma unroll
    for (int i = 0; i < 8; ++i)
      ob[(size_t)(d0 + i + h * 8) * NN + n0 + j * 16 + (lane & 15)] = acc[j][i];
}

// ---- Host driver ------------------------------------------------------------
extern "C" void kernel_launch(void* const* d_in, const int* in_sizes, int n_in,
                              void* d_out, int out_size, void* d_ws,
                              size_t ws_size, hipStream_t stream) {
  const float* x = (const float*)d_in[0];      // [32,512,4096] f32
  const float* bases = (const float*)d_in[1];  // [32,512,64]  f32
  float* out = (float*)d_out;                  // [32,512,4096] f32

  // Workspace carve-up (~183 MB total; all segments 256B-aligned by size).
  __bf16* XT = (__bf16*)d_ws;                              // B*N*D bf16 (128MB)
  float* coef = (float*)(XT + (size_t)BB * NN * DD);       // B*N*R f32  (32MB)
  __bf16* coefT = (__bf16*)(coef + (size_t)BB * NN * RR);  // B*R*N bf16 (16MB)
  float* basesW = (float*)(coefT + (size_t)BB * RR * NN);  // B*D*R f32  (4MB)
  __bf16* basesT = (__bf16*)(basesW + (size_t)BB * DD * RR); // B*R*D bf16 (2MB)
  __bf16* gramB = basesT + (size_t)BB * RR * DD;           // B*64*64 bf16
  __bf16* gramC = gramB + (size_t)BB * RR * RR;            // B*64*64 bf16

  const dim3 blk(256);
  // Prep: XT = bf16(x^T); working copies of bases.
  k_prep_x<<<dim3(NN / 64, DD / 64, BB), blk, 0, stream>>>(x, XT);
  k_prep_bases<<<dim3(BB), blk, 0, stream>>>(bases, basesW, basesT);

  // coef = softmax(x^T @ bases)  (INV_T = 1)
  k_coef_update<<<dim3(NN / 128, BB), blk, 0, stream>>>(XT, basesT, gramB, coef,
                                                        coefT, 0);
  // 7 multiplicative-update iterations
  for (int it = 0; it < 7; ++it) {
    k_gram<<<dim3(BB), blk, 0, stream>>>(basesT, DD, gramB);
    k_coef_update<<<dim3(NN / 128, BB), blk, 0, stream>>>(XT, basesT, gramB,
                                                          coef, coefT, 1);
    k_gram<<<dim3(BB), blk, 0, stream>>>(coefT, NN, gramC);
    k_bases_update<<<dim3(DD / 128, BB), blk, 0, stream>>>(x, coefT, gramC,
                                                           basesW, basesT);
  }
  // Final compute_coef (same multiplicative coef update with frozen bases)
  k_gram<<<dim3(BB), blk, 0, stream>>>(basesT, DD, gramB);
  k_coef_update<<<dim3(NN / 128, BB), blk, 0, stream>>>(XT, basesT, gramB, coef,
                                                        coefT, 1);
  // out = bases @ coef^T
  k_recon<<<dim3(NN / 64, DD / 128, BB), blk, 0, stream>>>(basesW, coef, out);
}
